// TiledFlashAttention_62380105007293
// MI455X (gfx1250) — matrix operand order repaired and checked by reading, not compiled, not run on hardware
//
#include <hip/hip_runtime.h>
#include <hip/hip_bf16.h>

typedef __attribute__((ext_vector_type(16))) __bf16 bfx16;
typedef __attribute__((ext_vector_type(8)))  __bf16 bfx8;
typedef __attribute__((ext_vector_type(8)))  float  fx8;
typedef __attribute__((ext_vector_type(4)))  unsigned int u32x4;
typedef __attribute__((ext_vector_type(8)))  int i32x8;
typedef __attribute__((ext_vector_type(4)))  int i32x4;

#define WPB 8               // waves per block (wave32)
#define TPB (WPB * 32)

static constexpr int Bc  = 2;
static constexpr int Sc  = 2048;
static constexpr int Dc  = 1024;
static constexpr int Hc  = 16;
static constexpr int HDc = 64;

__device__ __forceinline__ fx8 wmma_bf16(bfx16 a, bfx16 b, fx8 c) {
  // D = A(16x32 bf16) * B(32x16 bf16) + C(16x16 f32)
  return __builtin_amdgcn_wmma_f32_16x16x32_bf16(false, a, false, b, (short)0, c,
                                                 false, false);
}

// A-operand fragment (16x32 bf16): lane holds row M=lane&15,
// elements 0-15 -> K = 2*kb..2*kb+15 (contiguous),
// kb = 8*(lane>>4). Two 16-byte loads (global or LDS).
__device__ __forceinline__ bfx16 load_a_tile(const __bf16* rowbase, int kb) {
  bfx8 lo = *(const bfx8*)(rowbase + 2 * kb);
  bfx8 hi = *(const bfx8*)(rowbase + 2 * kb + 8);
  bfx16 r;
#pragma unroll
  for (int i = 0; i < 8; ++i) { r[i] = lo[i]; r[i + 8] = hi[i]; }
  return r;
}

// ---------------- Tensor Data Mover helpers ----------------
// D# packing per CDNA5 ISA (2-D tile, bf16 data).

__device__ __forceinline__ u32x4 tdm_group0(unsigned lds_byte_off, const void* gaddr) {
  unsigned long long ga = (unsigned long long)(uintptr_t)gaddr;
  u32x4 g;
  g[0] = 1u;                                                   // count=1, user D#
  g[1] = lds_byte_off;                                         // lds_addr [63:32]
  g[2] = (unsigned)(ga & 0xffffffffu);                         // global_addr lo
  g[3] = (unsigned)((ga >> 32) & 0x1ffffffu) | (2u << 30);     // addr hi | type=2
  return g;
}

__device__ __forceinline__ i32x8 tdm_group1(unsigned tensor_d0, unsigned tensor_d1,
                                            unsigned tile_d0, unsigned tile_d1,
                                            unsigned long long stride0) {
  i32x8 g;
  g[0] = (int)(1u << 16);                                      // data_size=1 -> 2B
  g[1] = (int)((tensor_d0 & 0xffffu) << 16);                   // tensor_dim0 lo
  g[2] = (int)((tensor_d0 >> 16) | ((tensor_d1 & 0xffffu) << 16));
  g[3] = (int)((tensor_d1 >> 16) | ((tile_d0 & 0xffffu) << 16));
  g[4] = (int)(tile_d1 & 0xffffu);                             // tile_dim1, tile_dim2=0
  g[5] = (int)(stride0 & 0xffffffffu);                         // tensor_dim0_stride lo
  g[6] = (int)((stride0 >> 32) & 0xffffu);                     // stride hi | dim1_stride=0
  g[7] = 0;
  return g;
}

// Async 2-D tile load: global (rows of tile_d0 elems, row stride stride0 elems)
// -> LDS packed rows (tile_d0*2 bytes each). bf16 elements.
__device__ __forceinline__ void tdm_load_2d(void* lds, const void* g,
                                            unsigned tile_d0, unsigned tile_d1,
                                            unsigned long long stride0) {
#if __has_builtin(__builtin_amdgcn_tensor_load_to_lds)
  u32x4 g0 = tdm_group0((unsigned)(uintptr_t)lds, g);
  i32x8 g1 = tdm_group1((unsigned)stride0, 1u << 20, tile_d0, tile_d1, stride0);
  i32x4 z4 = {};
  i32x8 z8 = {};
  // clang-23 / therock-10.0 form: (g0, g1, g2, g3, g4, cpol)
  __builtin_amdgcn_tensor_load_to_lds(g0, g1, z4, z4, z8, 0);
#else
  // fallback: cooperative copy by the issuing wave
  const __bf16* src = (const __bf16*)g;
  __bf16* dst = (__bf16*)lds;
  const int lane = threadIdx.x & 31;
  for (unsigned r = 0; r < tile_d1; ++r)
    for (unsigned c = (unsigned)lane; c < tile_d0; c += 32)
      dst[r * tile_d0 + c] = src[(size_t)r * stride0 + c];
#endif
}

__device__ __forceinline__ void tdm_wait() {
#if __has_builtin(__builtin_amdgcn_s_wait_tensorcnt)
  __builtin_amdgcn_s_wait_tensorcnt(0);
#else
  asm volatile("s_wait_tensorcnt 0x0" ::: "memory");
#endif
}

// ---------------- conversion kernels ----------------

__global__ __launch_bounds__(256) void cvt_bf16_kernel(const float* __restrict__ in,
                                                       __bf16* __restrict__ out, int n) {
  int i = blockIdx.x * blockDim.x + threadIdx.x;
  if (i < n) out[i] = (__bf16)in[i];
}

// in [K][N] row-major  ->  out [N][K] (B-operand friendly)
__global__ __launch_bounds__(256) void cvt_bf16_T_kernel(const float* __restrict__ in,
                                                         __bf16* __restrict__ out,
                                                         int K, int N) {
  int i = blockIdx.x * blockDim.x + threadIdx.x;
  if (i < K * N) {
    int k = i / N, n = i % N;
    out[(size_t)n * K + k] = (__bf16)in[i];
  }
}

// ---------------- block-tiled GEMM with TDM-staged B ----------------
// Block = 8 waves = 128(M) x 64(N) tile. The 64x32 B tile per K-step is shared
// by all waves: staged into LDS by the TDM (double-buffered), consumed via
// ds_load_b128 fragments. A rows stream straight from global (per-wave unique).

template <int EPI>  // 0: QKV scatter epilogue, 1: plain f32 store
__global__ __launch_bounds__(TPB) void gemm_kernel(const __bf16* __restrict__ A,
                                                   const __bf16* __restrict__ BT,
                                                   int K, int numN64,
                                                   __bf16* __restrict__ Qb,
                                                   __bf16* __restrict__ Kb,
                                                   __bf16* __restrict__ VTb,
                                                   float* __restrict__ Out) {
  __shared__ __bf16 btile[2][64 * 32];     // [buf][row N][k] : 4KB each
  const int lane = threadIdx.x & 31, wave = threadIdx.x >> 5;
  const int hi = lane >> 4, n15 = lane & 15;
  const int kb = 8 * hi, kb2 = 16 * hi;
  const int bm = blockIdx.x / numN64, bn = blockIdx.x % numN64;
  const int tm = bm * WPB + wave;          // 16-row M tile of this wave
  const int tn = bn;                       // 64-wide N tile of this block

  const __bf16* arow = A + ((size_t)tm * 16 + n15) * K;
  const __bf16* bsrc = BT + (size_t)(tn * 64) * K;

  if (wave == 0) {
    tdm_load_2d(&btile[0][0], bsrc, 32, 64, (unsigned long long)K);
    tdm_wait();
  }
  __syncthreads();

  fx8 acc0 = (fx8){}, acc1 = (fx8){}, acc2 = (fx8){}, acc3 = (fx8){};
  for (int k = 0; k < K; k += 32) {
    const int cur = (k >> 5) & 1, nxt = cur ^ 1;
    const bool more = (k + 32) < K;
    if (wave == 0 && more)
      tdm_load_2d(&btile[nxt][0], bsrc + (k + 32), 32, 64, (unsigned long long)K);

    __builtin_prefetch(arow + k + 256, 0, 3);
    bfx16 a = load_a_tile(arow + k, kb);
    const __bf16* bl = &btile[cur][0];
    bfx16 b0 = *(const bfx16*)(bl + (0  + n15) * 32 + kb2);
    bfx16 b1 = *(const bfx16*)(bl + (16 + n15) * 32 + kb2);
    bfx16 b2 = *(const bfx16*)(bl + (32 + n15) * 32 + kb2);
    bfx16 b3 = *(const bfx16*)(bl + (48 + n15) * 32 + kb2);
    acc0 = wmma_bf16(a, b0, acc0);
    acc1 = wmma_bf16(a, b1, acc1);
    acc2 = wmma_bf16(a, b2, acc2);
    acc3 = wmma_bf16(a, b3, acc3);

    if (wave == 0 && more) tdm_wait();
    __syncthreads();
  }

  fx8 accs[4] = {acc0, acc1, acc2, acc3};
  if (EPI == 0) {
    const int gbase = tn * 64;
    const int part  = gbase / Dc;          // 64-tile never straddles 1024 boundary
    const int h     = (gbase % Dc) >> 6;
#pragma unroll
    for (int t = 0; t < 4; ++t) {
      const int e = t * 16 + n15;
#pragma unroll
      for (int v = 0; v < 8; ++v) {
        const int mrow = tm * 16 + v + 8 * hi;
        const int b_ = mrow / Sc, s = mrow % Sc;
        const size_t bh = (size_t)(b_ * Hc + h);
        const float val = accs[t][v];
        if (part == 0)      Qb[(bh * Sc + s) * HDc + e] = (__bf16)(val * 0.125f);
        else if (part == 1) Kb[(bh * Sc + s) * HDc + e] = (__bf16)val;
        else                VTb[(bh * HDc + e) * Sc + s] = (__bf16)val;
      }
    }
  } else {
    const int N = numN64 * 64;
#pragma unroll
    for (int t = 0; t < 4; ++t) {
      const int gcol = tn * 64 + t * 16 + n15;
#pragma unroll
      for (int v = 0; v < 8; ++v) {
        const int mrow = tm * 16 + v + 8 * hi;
        Out[(size_t)mrow * N + gcol] = accs[t][v];
      }
    }
  }
}

// ---------------- flash attention (causal, online softmax) ----------------
// Block = one (b,h) and 8 consecutive 16-query tiles (128 queries). All waves
// share the K/V key-blocks of 32, staged into LDS by the TDM (double-buffered).
// Waves past their causal range see fully-masked blocks: exp->0, scale=1 (no-op),
// so the loop count is block-uniform and barriers never diverge.

__global__ __launch_bounds__(TPB) void attn_kernel(const __bf16* __restrict__ Qb,
                                                   const __bf16* __restrict__ Kb,
                                                   const __bf16* __restrict__ VTb,
                                                   __bf16* __restrict__ Ob) {
  __shared__ __bf16 ktile[2][32 * 64];     // [buf][key][hd]      4KB each
  __shared__ __bf16 vtile[2][64 * 32];     // [buf][hd][key]      4KB each
  __shared__ __bf16 plds[WPB][16 * 32];    // per-wave P transpose

  const int lane = threadIdx.x & 31, wave = threadIdx.x >> 5;
  const int hi = lane >> 4, n15 = lane & 15;
  const int kb = 8 * hi, kb2 = 16 * hi;
  const int bh = blockIdx.x / (Sc / 16 / WPB);     // 0..B*H-1
  const int qg = blockIdx.x % (Sc / 16 / WPB);     // query-tile group
  const int qt = qg * WPB + wave;                  // this wave's 16-query tile

  const __bf16* kbase = Kb  + (size_t)bh * Sc * HDc;
  const __bf16* vbase = VTb + (size_t)bh * HDc * Sc;

  // Q tile (loop invariant): A-operand, two K=32 chunks over hd=64
  const __bf16* qrow = Qb + ((size_t)bh * Sc + qt * 16 + n15) * HDc;
  bfx16 a0 = load_a_tile(qrow, kb);
  bfx16 a1 = load_a_tile(qrow + 32, kb);

  float m[8], l[8];
  fx8 acc[4];
#pragma unroll
  for (int v = 0; v < 8; ++v) { m[v] = -3.0e38f; l[v] = 0.0f; }
#pragma unroll
  for (int t = 0; t < 4; ++t) acc[t] = (fx8){};

  const int kend_blk = (qg * WPB + WPB) * 16;      // max key range in this block

  if (wave == 0) {
    tdm_load_2d(&ktile[0][0], kbase, 32 * 64, 1, 32 * 64);        // 32 contig rows
    tdm_load_2d(&vtile[0][0], vbase, 32, 64, (unsigned long long)Sc);
    tdm_wait();
  }
  __syncthreads();

  for (int j = 0; j < kend_blk; j += 32) {
    const int cur = (j >> 5) & 1, nxt = cur ^ 1;
    const bool more = (j + 32) < kend_blk;
    if (wave == 0 && more) {
      tdm_load_2d(&ktile[nxt][0], kbase + (size_t)(j + 32) * HDc, 32 * 64, 1, 32 * 64);
      tdm_load_2d(&vtile[nxt][0], vbase + (j + 32), 32, 64, (unsigned long long)Sc);
    }

    const __bf16* kl = &ktile[cur][0];
    const __bf16* vl = &vtile[cur][0];

    // scores: S = Q @ K^T  (LDS K rows are B-operand layout)
    fx8 s0 = (fx8){}, s1 = (fx8){};
    s0 = wmma_bf16(a0, *(const bfx16*)(kl + (n15)*HDc + kb2), s0);
    s0 = wmma_bf16(a1, *(const bfx16*)(kl + (n15)*HDc + 32 + kb2), s0);
    s1 = wmma_bf16(a0, *(const bfx16*)(kl + (16 + n15) * HDc + kb2), s1);
    s1 = wmma_bf16(a1, *(const bfx16*)(kl + (16 + n15) * HDc + 32 + kb2), s1);

    // causal mask + online softmax (rows live in lane halves of 16)
    float p0[8], p1[8];
#pragma unroll
    for (int v = 0; v < 8; ++v) {
      const int q = qt * 16 + v + 8 * hi;
      float sv0 = (j + n15      <= q) ? s0[v] : -3.0e38f;
      float sv1 = (j + 16 + n15 <= q) ? s1[v] : -3.0e38f;
      float rmax = fmaxf(sv0, sv1);
#pragma unroll
      for (int o = 8; o >= 1; o >>= 1) rmax = fmaxf(rmax, __shfl_xor(rmax, o, 16));
      const float mn = fmaxf(m[v], rmax);
      const float sc = __expf(m[v] - mn);
      const float e0 = __expf(sv0 - mn);
      const float e1 = __expf(sv1 - mn);
      float rs = e0 + e1;
#pragma unroll
      for (int o = 8; o >= 1; o >>= 1) rs += __shfl_xor(rs, o, 16);
      l[v] = l[v] * sc + rs;
      m[v] = mn;
      p0[v] = e0;
      p1[v] = e1;
#pragma unroll
      for (int t = 0; t < 4; ++t) acc[t][v] *= sc;
    }

    // P: C layout -> A layout via per-wave LDS transpose
    __bf16* pl = &plds[wave][0];
#pragma unroll
    for (int v = 0; v < 8; ++v) {
      const int r = v + 8 * hi;
      pl[r * 32 + n15]      = (__bf16)p0[v];
      pl[r * 32 + 16 + n15] = (__bf16)p1[v];
    }
    asm volatile("s_wait_dscnt 0" ::: "memory");
    bfx16 pa = load_a_tile(pl + n15 * 32, kb);

    // O += P @ V  (LDS VT rows are B-operand layout)
#pragma unroll
    for (int t = 0; t < 4; ++t)
      acc[t] = wmma_bf16(pa, *(const bfx16*)(vl + (t * 16 + n15) * 32 + kb2), acc[t]);

    if (wave == 0 && more) tdm_wait();
    __syncthreads();
  }

  // normalize + store O in [B][S][D] bf16 (A-operand friendly for out-proj)
  const int b_ = bh >> 4;
  const int h  = bh & 15;
#pragma unroll
  for (int t = 0; t < 4; ++t) {
#pragma unroll
    for (int v = 0; v < 8; ++v) {
      const int q = qt * 16 + v + 8 * hi;
      Ob[((size_t)b_ * Sc + q) * Dc + h * HDc + t * 16 + n15] =
          (__bf16)(acc[t][v] / l[v]);
    }
  }
}

// ---------------- host launch ----------------

extern "C" void kernel_launch(void* const* d_in, const int* in_sizes, int n_in,
                              void* d_out, int out_size, void* d_ws, size_t ws_size,
                              hipStream_t stream) {
  (void)in_sizes; (void)n_in; (void)out_size; (void)ws_size;
  const float* x     = (const float*)d_in[0];   // [2,2048,1024]
  const float* w_qkv = (const float*)d_in[1];   // [1024,3072]
  const float* w_out = (const float*)d_in[2];   // [1024,1024]
  float* out = (float*)d_out;                   // [2,2048,1024] f32

  char* ws = (char*)d_ws;
  size_t off = 0;
  auto carve = [&](size_t elems) -> __bf16* {
    __bf16* p = (__bf16*)(ws + off);
    off += elems * sizeof(__bf16);
    off = (off + 255) & ~(size_t)255;
    return p;
  };
  const size_t nX  = (size_t)Bc * Sc * Dc;        // 4,194,304
  const size_t nWq = (size_t)Dc * 3 * Dc;         // 3,145,728
  const size_t nWo = (size_t)Dc * Dc;             // 1,048,576
  const size_t nHD = (size_t)Bc * Hc * Sc * HDc;  // 4,194,304

  __bf16* xb    = carve(nX);
  __bf16* wqkvT = carve(nWq);
  __bf16* woutT = carve(nWo);
  __bf16* Qb    = carve(nHD);
  __bf16* Kb    = carve(nHD);
  __bf16* VTb   = carve(nHD);
  __bf16* Ob    = carve(nX);

  // 1) precision conversion (+ weight transpose to [N][K])
  cvt_bf16_kernel<<<(int)(nX / 256), 256, 0, stream>>>(x, xb, (int)nX);
  cvt_bf16_T_kernel<<<(int)(nWq / 256), 256, 0, stream>>>(w_qkv, wqkvT, Dc, 3 * Dc);
  cvt_bf16_T_kernel<<<(int)(nWo / 256), 256, 0, stream>>>(w_out, woutT, Dc, Dc);

  // 2) QKV projection: (4096/128) x (3072/64) blocks
  gemm_kernel<0><<<(Bc * Sc / 128) * ((3 * Dc) / 64), TPB, 0, stream>>>(
      xb, wqkvT, Dc, (3 * Dc) / 64, Qb, Kb, VTb, nullptr);

  // 3) causal flash attention: B*H * (S/16/8) blocks
  attn_kernel<<<Bc * Hc * (Sc / 16 / WPB), TPB, 0, stream>>>(Qb, Kb, VTb, Ob);

  // 4) output projection -> f32: (4096/128) x (1024/64) blocks
  gemm_kernel<1><<<(Bc * Sc / 128) * (Dc / 64), TPB, 0, stream>>>(
      Ob, woutT, Dc, Dc / 64, nullptr, nullptr, nullptr, out);
}
